// CrossAttention_15779709846265
// MI455X (gfx1250) — matrix operand order repaired and checked by reading, not compiled, not run on hardware
//
#include <hip/hip_runtime.h>
#include <hip/hip_bf16.h>

// ---------------------------------------------------------------------------
// CrossAttention for MI455X (gfx1250), wave32 + v_wmma_f32_16x16x32_bf16
// + Tensor Data Mover (tensor_load_to_lds) for attention K-tile staging.
//
// Reference quirks exploited: V == K (kv[:,:,0] twice), so only W_kv[:1024]
// is needed and the attention kernel reuses the staged K tile for both
// S = Q K^T and O += P V.  softmax scale folded into Q projection.
// ---------------------------------------------------------------------------

typedef __bf16 bf16_t;
typedef __attribute__((ext_vector_type(16))) __bf16 v16bf;
typedef __attribute__((ext_vector_type(8)))  __bf16 v8bf;
typedef __attribute__((ext_vector_type(8)))  float  v8f;
typedef __attribute__((ext_vector_type(4)))  unsigned int u32x4;
typedef __attribute__((ext_vector_type(8)))  unsigned int u32x8;

#define DIM_C      1024
#define HEADS_C    16
#define HEAD_DIM_C 64
#define BATCH_C    2
#define NQ_C       2048
#define MK_C       2048

__device__ __forceinline__ v8f wmma_bf16(v16bf a, v16bf b, v8f c) {
  // D = A(16x32) * B(32x16) + C(16x16 f32)
  return __builtin_amdgcn_wmma_f32_16x16x32_bf16(
      /*neg_a=*/false, a, /*neg_b=*/false, b,
      /*c_mod=*/(short)0, c, /*reuse_a=*/false, /*reuse_b=*/false);
}

// A-fragment (16x32 bf16): lane holds row = lane&15; K elements are
// K = (lane>>4)*16 + {0..15}.
// Caller passes p = &A[row*lda + k0 + (lane>>4)*16]; chunks at p[0..7], p[8..15].
__device__ __forceinline__ v16bf loadA_bf16(const bf16_t* p) {
  v8bf lo = *(const v8bf*)p;
  v8bf hi = *(const v8bf*)(p + 8);
  v16bf a;
#pragma unroll
  for (int i = 0; i < 8; ++i) { a[i] = lo[i]; a[i + 8] = hi[i]; }
  return a;
}

// ---------------------------------------------------------------------------
// fp32 -> bf16 elementwise convert
// ---------------------------------------------------------------------------
__global__ void cvt_f32_bf16(const float* __restrict__ src,
                             bf16_t* __restrict__ dst, int n) {
  int i = blockIdx.x * blockDim.x + threadIdx.x;
  if (i < n) dst[i] = (bf16_t)src[i];
}

// ---------------------------------------------------------------------------
// C(bf16)[M x N] = scale * A(bf16)[M x K] @ W(bf16)[N x K]^T
// 256 threads = 8 waves; wave grid 4x2 -> 128x64 tile per WG; 32x32 per wave.
// ---------------------------------------------------------------------------
__global__ __launch_bounds__(256)
void gemm_bf16_wmma(const bf16_t* __restrict__ A, const bf16_t* __restrict__ W,
                    bf16_t* __restrict__ C, int M, int N, int K, float scale) {
  const int lane = threadIdx.x & 31;
  const int wave = threadIdx.x >> 5;
  const int r  = lane & 15;
  const int hi = lane >> 4;
  const int row0 = blockIdx.x * 128 + (wave >> 1) * 32;
  const int col0 = blockIdx.y * 64  + (wave & 1) * 32;

  v8f acc00 = {}, acc01 = {}, acc10 = {}, acc11 = {};

  const bf16_t* pa0 = A + (size_t)(row0 + r)      * K + hi * 16;
  const bf16_t* pa1 = A + (size_t)(row0 + 16 + r) * K + hi * 16;
  const bf16_t* pb0 = W + (size_t)(col0 + r)      * K + hi * 16;
  const bf16_t* pb1 = W + (size_t)(col0 + 16 + r) * K + hi * 16;

  for (int k0 = 0; k0 < K; k0 += 32) {
    __builtin_prefetch((const void*)(pa0 + k0 + 128), 0, 1);
    __builtin_prefetch((const void*)(pb0 + k0 + 128), 0, 1);
    v16bf a0 = loadA_bf16(pa0 + k0);
    v16bf a1 = loadA_bf16(pa1 + k0);
    v16bf b0 = *(const v16bf*)(pb0 + k0);
    v16bf b1 = *(const v16bf*)(pb1 + k0);
    acc00 = wmma_bf16(a0, b0, acc00);
    acc01 = wmma_bf16(a0, b1, acc01);
    acc10 = wmma_bf16(a1, b0, acc10);
    acc11 = wmma_bf16(a1, b1, acc11);
  }

#pragma unroll
  for (int rr = 0; rr < 8; ++rr) {
    int rA = row0 + hi * 8 + rr;
    int rB = rA + 16;
    C[(size_t)rA * N + col0 + r]      = (bf16_t)(acc00[rr] * scale);
    C[(size_t)rA * N + col0 + 16 + r] = (bf16_t)(acc01[rr] * scale);
    C[(size_t)rB * N + col0 + r]      = (bf16_t)(acc10[rr] * scale);
    C[(size_t)rB * N + col0 + 16 + r] = (bf16_t)(acc11[rr] * scale);
  }
}

// ---------------------------------------------------------------------------
// out(f32)[M x N] = A(bf16)[M x K] @ W(bf16)[N x K]^T + bias[N]
// ---------------------------------------------------------------------------
__global__ __launch_bounds__(256)
void gemm_bf16A_bias_wmma(const bf16_t* __restrict__ A,
                          const bf16_t* __restrict__ W,
                          const float* __restrict__ bias,
                          float* __restrict__ C, int M, int N, int K) {
  const int lane = threadIdx.x & 31;
  const int wave = threadIdx.x >> 5;
  const int r  = lane & 15;
  const int hi = lane >> 4;
  const int row0 = blockIdx.x * 128 + (wave >> 1) * 32;
  const int col0 = blockIdx.y * 64  + (wave & 1) * 32;

  v8f acc00 = {}, acc01 = {}, acc10 = {}, acc11 = {};

  const bf16_t* pa0 = A + (size_t)(row0 + r)      * K + hi * 16;
  const bf16_t* pa1 = A + (size_t)(row0 + 16 + r) * K + hi * 16;
  const bf16_t* pb0 = W + (size_t)(col0 + r)      * K + hi * 16;
  const bf16_t* pb1 = W + (size_t)(col0 + 16 + r) * K + hi * 16;

  for (int k0 = 0; k0 < K; k0 += 32) {
    __builtin_prefetch((const void*)(pa0 + k0 + 128), 0, 1);
    __builtin_prefetch((const void*)(pb0 + k0 + 128), 0, 1);
    v16bf a0 = loadA_bf16(pa0 + k0);
    v16bf a1 = loadA_bf16(pa1 + k0);
    v16bf b0 = *(const v16bf*)(pb0 + k0);
    v16bf b1 = *(const v16bf*)(pb1 + k0);
    acc00 = wmma_bf16(a0, b0, acc00);
    acc01 = wmma_bf16(a0, b1, acc01);
    acc10 = wmma_bf16(a1, b0, acc10);
    acc11 = wmma_bf16(a1, b1, acc11);
  }

  float bia = bias[col0 + r];
  float bib = bias[col0 + 16 + r];
#pragma unroll
  for (int rr = 0; rr < 8; ++rr) {
    int rA = row0 + hi * 8 + rr;
    int rB = rA + 16;
    C[(size_t)rA * N + col0 + r]      = acc00[rr] + bia;
    C[(size_t)rA * N + col0 + 16 + r] = acc01[rr] + bib;
    C[(size_t)rB * N + col0 + r]      = acc10[rr] + bia;
    C[(size_t)rB * N + col0 + 16 + r] = acc11[rr] + bib;
  }
}

// ---------------------------------------------------------------------------
// Flash attention, V == K.  Grid: (N/64, B*HEADS).  128 threads = 4 waves;
// wave w owns query rows [q0 + w*16, +16).  Keys streamed in blocks of 32,
// staged into LDS by the Tensor Data Mover (double-buffered, TENSORcnt).
// ---------------------------------------------------------------------------
__global__ __launch_bounds__(128)
void attn_wmma(const bf16_t* __restrict__ Q, const bf16_t* __restrict__ Km,
               bf16_t* __restrict__ O) {
  // TDM writes each 64-elem (128 B) row then pads 8 DWORDs -> row stride 80
  // elements (160 B), keeping 32 B alignment for v16bf reloads.
  __shared__ bf16_t Kblk[2][32][80];   // double-buffered 32 keys x 64 dims
  __shared__ bf16_t Pbuf[4][16][40];   // per-wave P relayout buffer (16x32)

  const int lane = threadIdx.x & 31;
  const int wave = threadIdx.x >> 5;
  const int r  = lane & 15;
  const int hi = lane >> 4;

  const int bh = blockIdx.y;               // b*HEADS + h
  const int b  = bh >> 4;
  const int h  = bh & 15;
  const size_t base = (size_t)b * NQ_C * DIM_C + (size_t)h * HEAD_DIM_C;
  const int q0 = blockIdx.x * 64 + wave * 16;

  // Q fragments (A layout), head-dim K-chunks 0..31 and 32..63 (scale folded)
  const bf16_t* qrow = Q + base + (size_t)(q0 + r) * DIM_C;
  v16bf qa0 = loadA_bf16(qrow + 0  + hi * 16);
  v16bf qa1 = loadA_bf16(qrow + 32 + hi * 16);

  float rmax[8], rsum[8];
  v8f o0 = {}, o1 = {}, o2 = {}, o3 = {};
#pragma unroll
  for (int i = 0; i < 8; ++i) { rmax[i] = -3.0e30f; rsum[i] = 0.0f; }

  // ---- TDM descriptor issue: 2D tile 32 keys x 64 bf16, stride DIM --------
  auto tdm_issue = [&](int buf, int j0) {
    unsigned long long ga =
        (unsigned long long)(uintptr_t)(Km + base + (size_t)j0 * DIM_C);
    unsigned lds = (unsigned)(uintptr_t)(&Kblk[buf][0][0]);
    u32x4 g0;
    g0[0] = 1u;                                   // count=1, no gather
    g0[1] = lds;                                  // lds_addr (bytes)
    g0[2] = (unsigned)(ga & 0xffffffffu);         // global_addr[31:0]
    g0[3] = (unsigned)((ga >> 32) & 0x1ffffffu)   // global_addr[56:32]
          | (2u << 30);                           // type = 2 (image)
    u32x8 g1;
    g1[0] = (1u << 16)        // data_size = 2 bytes
          | (1u << 20)        // pad_enable
          | (4u << 22)        // pad_interval: 32 DWORDs (=128 B row)
          | (7u << 25);       // pad_amount: 8 DWORDs (=16 elements)
    g1[1] = (64u << 16);      // tensor_dim0 = 64 (low 16 bits @ [31:16])
    g1[2] = (32u << 16);      // tensor_dim1 = 32 (low 16 bits @ [31:16])
    g1[3] = (64u << 16);      // tile_dim0 = 64 @ [127:112]
    g1[4] = 32u;              // tile_dim1 = 32 @ [143:128]; tile_dim2 = 0
    g1[5] = (unsigned)DIM_C;  // tensor_dim0_stride = 1024 (low 32 of 48)
    g1[6] = 0u;               // stride hi, tensor_dim1_stride lo (unused, 2D)
    g1[7] = 0u;
    // 2D tensor: SGPR groups 2/3 are NULL -> 2-operand form
    asm volatile("tensor_load_to_lds %0, %1" :: "s"(g0), "s"(g1) : "memory");
  };

  if (wave == 0) tdm_issue(0, 0);

  for (int jb = 0; jb < MK_C / 32; ++jb) {
    const int cur = jb & 1;
    if (wave == 0) __builtin_amdgcn_s_wait_tensorcnt(0);  // cur buffer ready
    __syncthreads();
    // prefetch next key block into the other buffer (overlaps compute)
    if (wave == 0 && jb + 1 < MK_C / 32) tdm_issue(cur ^ 1, (jb + 1) * 32);

    // ---- S = Q * K^T : two 16-key column tiles, K-dim = head_dim (2 steps)
    v8f s0, s1;
    {
      v16bf bk0 = *(const v16bf*)(&Kblk[cur][r][hi * 16]);
      v16bf bk1 = *(const v16bf*)(&Kblk[cur][r][32 + hi * 16]);
      v8f a = {};
      a = wmma_bf16(qa0, bk0, a);
      a = wmma_bf16(qa1, bk1, a);
      s0 = a;
    }
    {
      v16bf bk0 = *(const v16bf*)(&Kblk[cur][16 + r][hi * 16]);
      v16bf bk1 = *(const v16bf*)(&Kblk[cur][16 + r][32 + hi * 16]);
      v8f a = {};
      a = wmma_bf16(qa0, bk0, a);
      a = wmma_bf16(qa1, bk1, a);
      s1 = a;
    }

    // ---- streaming softmax; row reductions stay within 16-lane halves
    float p0[8], p1[8];
#pragma unroll
    for (int rr = 0; rr < 8; ++rr) {
      float m = fmaxf(s0[rr], s1[rr]);
#pragma unroll
      for (int mk = 1; mk < 16; mk <<= 1)
        m = fmaxf(m, __shfl_xor(m, mk, 32));
      float mnew = fmaxf(m, rmax[rr]);
      float e0 = __expf(s0[rr] - mnew);
      float e1 = __expf(s1[rr] - mnew);
      p0[rr] = e0; p1[rr] = e1;
      float ps = e0 + e1;
#pragma unroll
      for (int mk = 1; mk < 16; mk <<= 1)
        ps += __shfl_xor(ps, mk, 32);
      float corr = __expf(rmax[rr] - mnew);
      rsum[rr] = rsum[rr] * corr + ps;
      rmax[rr] = mnew;
      o0[rr] *= corr; o1[rr] *= corr; o2[rr] *= corr; o3[rr] *= corr;
    }

    // ---- relayout P (C layout -> A layout) via per-wave LDS scratch
#pragma unroll
    for (int rr = 0; rr < 8; ++rr) {
      Pbuf[wave][hi * 8 + rr][r]      = (bf16_t)p0[rr];
      Pbuf[wave][hi * 8 + rr][16 + r] = (bf16_t)p1[rr];
    }
    asm volatile("s_wait_dscnt 0x0" ::: "memory");  // same-wave LDS RAW
    v16bf pa = loadA_bf16(&Pbuf[wave][r][hi * 16]);

    // ---- O += P * V  (V == K); B fragment: col = head-dim, K = key
    v16bf bv0, bv1, bv2, bv3;
#pragma unroll
    for (int e = 0; e < 16; ++e) {
      const bf16_t* krow = &Kblk[cur][hi * 16 + e][r];
      bv0[e] = krow[0];
      bv1[e] = krow[16];
      bv2[e] = krow[32];
      bv3[e] = krow[48];
    }
    o0 = wmma_bf16(pa, bv0, o0);
    o1 = wmma_bf16(pa, bv1, o1);
    o2 = wmma_bf16(pa, bv2, o2);
    o3 = wmma_bf16(pa, bv3, o3);
  }

  // ---- finalize: O /= rowsum, store bf16 into (B*N, DIM) at head offset
#pragma unroll
  for (int rr = 0; rr < 8; ++rr) {
    float inv = 1.0f / rsum[rr];
    size_t orow = base + (size_t)(q0 + hi * 8 + rr) * DIM_C;
    O[orow + 0  + r] = (bf16_t)(o0[rr] * inv);
    O[orow + 16 + r] = (bf16_t)(o1[rr] * inv);
    O[orow + 32 + r] = (bf16_t)(o2[rr] * inv);
    O[orow + 48 + r] = (bf16_t)(o3[rr] * inv);
  }
}

// ---------------------------------------------------------------------------
// launch
// ---------------------------------------------------------------------------
extern "C" void kernel_launch(void* const* d_in, const int* in_sizes, int n_in,
                              void* d_out, int out_size, void* d_ws, size_t ws_size,
                              hipStream_t stream) {
  const float* x    = (const float*)d_in[0];   // (B,N,DIM)
  const float* ctx  = (const float*)d_in[1];   // (B,M,DIM)
  const float* Wq   = (const float*)d_in[2];   // (DIM,DIM)
  const float* Wkv  = (const float*)d_in[3];   // (2*DIM,DIM); only first DIM rows used
  const float* Wp   = (const float*)d_in[4];   // (DIM,DIM)
  const float* bp   = (const float*)d_in[5];   // (DIM,)
  float* out = (float*)d_out;

  const int ROWS = BATCH_C * NQ_C;             // 4096
  const int WN   = DIM_C * DIM_C;              // 1M weight elements
  const int AN   = ROWS * DIM_C;               // 4M activation elements

  char* ws = (char*)d_ws;
  const size_t MB = 1024 * 1024;
  bf16_t* wq_b = (bf16_t*)(ws + 0 * MB);       // 2 MB
  bf16_t* wk_b = (bf16_t*)(ws + 2 * MB);       // 2 MB
  bf16_t* wp_b = (bf16_t*)(ws + 4 * MB);       // 2 MB
  bf16_t* x_b  = (bf16_t*)(ws + 6 * MB);       // 8 MB
  bf16_t* c_b  = (bf16_t*)(ws + 14 * MB);      // 8 MB
  bf16_t* Qb   = (bf16_t*)(ws + 22 * MB);      // 8 MB
  bf16_t* Kb   = (bf16_t*)(ws + 30 * MB);      // 8 MB
  bf16_t* Ob   = (bf16_t*)(ws + 38 * MB);      // 8 MB  (total 46 MB)

  // 1) one-time bf16 conversions (W_k = first DIM rows of W_kv; V == K)
  cvt_f32_bf16<<<WN / 256, 256, 0, stream>>>(Wq,  wq_b, WN);
  cvt_f32_bf16<<<WN / 256, 256, 0, stream>>>(Wkv, wk_b, WN);
  cvt_f32_bf16<<<WN / 256, 256, 0, stream>>>(Wp,  wp_b, WN);
  cvt_f32_bf16<<<AN / 256, 256, 0, stream>>>(x,   x_b,  AN);
  cvt_f32_bf16<<<AN / 256, 256, 0, stream>>>(ctx, c_b,  AN);

  // 2) projections: Q = scale * x @ Wq^T ; K = ctx @ Wk^T
  dim3 ggrid(ROWS / 128, DIM_C / 64);
  const float scale = 0.125f;                  // HEAD_DIM^-0.5
  gemm_bf16_wmma<<<ggrid, 256, 0, stream>>>(x_b, wq_b, Qb, ROWS, DIM_C, DIM_C, scale);
  gemm_bf16_wmma<<<ggrid, 256, 0, stream>>>(c_b, wk_b, Kb, ROWS, DIM_C, DIM_C, 1.0f);

  // 3) attention (V == K), TDM-staged K tiles
  dim3 agrid(NQ_C / 64, BATCH_C * HEADS_C);
  attn_wmma<<<agrid, 128, 0, stream>>>(Qb, Kb, Ob);

  // 4) output projection + bias
  gemm_bf16A_bias_wmma<<<ggrid, 256, 0, stream>>>(Ob, wp_b, bp, out, ROWS, DIM_C, DIM_C);
}
